// GraphNeuralNetwork_88742614270234
// MI455X (gfx1250) — compile-verified, matrix operand order checked by reading, not run on hardware
//
#include <hip/hip_runtime.h>

// GAT layer: h = xW^T+b ; scores = leaky(si+sj+ab) masked ; softmax ; attn@h ; +x ; ELU
// N=3072, C=256, H=8 heads, D=32 head_dim.

#define GN 3072
#define GC 256
#define GH 8
#define GD 32

typedef __attribute__((ext_vector_type(16))) _Float16 v16h;
typedef __attribute__((ext_vector_type(8)))  _Float16 v8h;
typedef __attribute__((ext_vector_type(4)))  _Float16 v4h;
typedef __attribute__((ext_vector_type(8)))  float    v8f;

typedef __attribute__((address_space(3))) char lds_char;

static __device__ __forceinline__ v16h cat16(v8h lo, v8h hi) {
  return __builtin_shufflevector(lo, hi, 0,1,2,3,4,5,6,7,8,9,10,11,12,13,14,15);
}

// 32-bit LDS byte offset of a generic pointer into shared memory
// (addrspacecast generic -> local, then truncate).
static __device__ __forceinline__ unsigned lds_off(const void* p) {
  return (unsigned)(uintptr_t)(lds_char*)(char*)(p);
}

// Async global->LDS copy (16 bytes per lane), tracked by ASYNCcnt.
static __device__ __forceinline__ void async_copy_b128(unsigned lds_byte_off,
                                                       const void* gaddr) {
  asm volatile("global_load_async_to_lds_b128 %0, %1, off"
               :: "v"(lds_byte_off), "v"(gaddr) : "memory");
}
static __device__ __forceinline__ void wait_asynccnt0() {
  asm volatile("s_wait_asynccnt 0x0" ::: "memory");
}

// ---------------------------------------------------------------------------
// Kernel 1: h = x @ W^T + bW   (f16 operands, f32 accumulate via WMMA)
// 64x64 output tile per workgroup, 4 waves, K-step 32.
// Outputs: h32[N][C] (f32) and hT16[head][d][N] (f16, K-contiguous for phase 3 B-frags)
// ---------------------------------------------------------------------------
__global__ __launch_bounds__(128)
void k_gemm_h(const float* __restrict__ x, const float* __restrict__ W,
              const float* __restrict__ bW,
              float* __restrict__ h32, _Float16* __restrict__ hT16) {
  __shared__ _Float16 Atile[64 * 32];   // x tile   [m][k]
  __shared__ _Float16 Wtile[64 * 32];   // W tile   [o][k]  (== B^T, exactly the frag layout)
  const int i0 = blockIdx.x * 64;
  const int o0 = blockIdx.y * 64;
  const int tid  = threadIdx.x;
  const int wv   = tid >> 5;
  const int lane = tid & 31;
  const int l15  = lane & 15;
  const int kof  = (lane < 16) ? 0 : 8;   // 16-bit fragment K layout split

  v8f acc[4] = {};

  const int lr = tid >> 1;          // loader row 0..63
  const int lc = (tid & 1) * 16;    // loader col 0 or 16

  for (int kk = 0; kk < GC; kk += 32) {
    // Stage A and W tiles into LDS as f16 (b128 global loads, b128 ds stores).
    const float4* ap = (const float4*)(x + (size_t)(i0 + lr) * GC + kk + lc);
    const float4* wp = (const float4*)(W + (size_t)(o0 + lr) * GC + kk + lc);
    alignas(16) _Float16 ta[16];
    alignas(16) _Float16 tw[16];
#pragma unroll
    for (int q = 0; q < 4; ++q) {
      float4 av = ap[q]; float4 wvv = wp[q];
      ta[q*4+0] = (_Float16)av.x;  ta[q*4+1] = (_Float16)av.y;
      ta[q*4+2] = (_Float16)av.z;  ta[q*4+3] = (_Float16)av.w;
      tw[q*4+0] = (_Float16)wvv.x; tw[q*4+1] = (_Float16)wvv.y;
      tw[q*4+2] = (_Float16)wvv.z; tw[q*4+3] = (_Float16)wvv.w;
    }
    *(v8h*)(Atile + lr * 32 + lc)     = *(v8h*)(ta);
    *(v8h*)(Atile + lr * 32 + lc + 8) = *(v8h*)(ta + 8);
    *(v8h*)(Wtile + lr * 32 + lc)     = *(v8h*)(tw);
    *(v8h*)(Wtile + lr * 32 + lc + 8) = *(v8h*)(tw + 8);
    __syncthreads();

    // A fragment: ds_read_b128 pairs per the 16-bit 16x32 layout.
    const int rowA = wv * 16 + l15;
    v8h a_lo = *(const v8h*)(Atile + rowA * 32 + kof);
    v8h a_hi = *(const v8h*)(Atile + rowA * 32 + kof + 16);
    v16h afrag = cat16(a_lo, a_hi);
#pragma unroll
    for (int ct = 0; ct < 4; ++ct) {
      const int ocol = ct * 16 + l15;
      v8h b_lo = *(const v8h*)(Wtile + ocol * 32 + kof);
      v8h b_hi = *(const v8h*)(Wtile + ocol * 32 + kof + 16);
      v16h bfrag = cat16(b_lo, b_hi);
      acc[ct] = __builtin_amdgcn_wmma_f32_16x16x32_f16(
          false, afrag, false, bfrag, (short)0, acc[ct], false, false);
    }
    __syncthreads();
  }

  // Epilogue: C/D layout -> lane = column, VGPR r = row (+8 for lanes 16..31).
  const int mbase = (lane < 16) ? 0 : 8;
#pragma unroll
  for (int ct = 0; ct < 4; ++ct) {
#pragma unroll
    for (int r = 0; r < 8; ++r) {
      const int i = i0 + wv * 16 + mbase + r;
      const int o = o0 + ct * 16 + l15;
      const float v = acc[ct][r] + bW[o];
      h32[(size_t)i * GC + o] = v;
      const int head = o >> 5, d = o & 31;
      hT16[((size_t)head * GD + d) * GN + i] = (_Float16)v;
    }
  }
}

// ---------------------------------------------------------------------------
// Kernel 2: per-(node,head) attention logit halves, stored [H][N] so the
// attention kernel's per-head preload is a contiguous async copy.
// ---------------------------------------------------------------------------
__global__ __launch_bounds__(256)
void k_scores(const float* __restrict__ h32, const float* __restrict__ a1,
              const float* __restrict__ a2,
              float* __restrict__ siT, float* __restrict__ sjT) {
  const int t = blockIdx.x * blockDim.x + threadIdx.x;  // n*H + head
  if (t >= GN * GH) return;
  const int n = t >> 3, head = t & 7;
  const float* hp = h32 + (size_t)n * GC + head * GD;
  float s1 = 0.f, s2 = 0.f;
#pragma unroll
  for (int d = 0; d < GD; ++d) {
    const float v = hp[d];
    s1 += v * a1[d];
    s2 += v * a2[d];
  }
  siT[head * GN + n] = s1;
  sjT[head * GN + n] = s2;
}

// ---------------------------------------------------------------------------
// Kernel 3: one workgroup = (head, 16-row block).
//   Preload: per-head sj row -> LDS via GLOBAL_LOAD_ASYNC_TO_LDS_B128.
//   Phase A: masked LeakyReLU scores, int4/float4 vectorized, rowwise max &
//            exp-sum via wave32 shfl; exp(s-m) stored f16 in LDS.
//   Phase B: P[16xN] @ V[NxD] via v_wmma_f32_16x16x32_f16, 8 waves =
//            2 D-tiles x 4 K-chunks, partials reduced in LDS.
//   Epilogue: *1/rowsum, + skip(x), ELU.
// ---------------------------------------------------------------------------
#define PROB_STRIDE (GN + 8)
#define SMEM_SJ     (GN * 4)
#define SMEM_FIX    (SMEM_SJ + 64 + 64 + 2 * 4 * 16 * 16 * 4)
#define SMEM_TOTAL  (SMEM_FIX + 16 * PROB_STRIDE * 2)

__global__ __launch_bounds__(256)
void k_attn(const float* __restrict__ x, const int* __restrict__ adj,
            const float* __restrict__ siT, const float* __restrict__ sjT,
            const _Float16* __restrict__ hT16, const float* __restrict__ ab_p,
            float* __restrict__ out) {
  extern __shared__ char smem[];
  float*     sjh      = (float*)smem;                      // [N]
  float*     rowscale = (float*)(smem + SMEM_SJ);          // [16]
  float*     si16     = rowscale + 16;                     // [16]
  float*     partial  = si16 + 16;                         // [2][4][16][16]
  _Float16*  probs    = (_Float16*)(smem + SMEM_FIX);      // [16][PROB_STRIDE]

  const int i0   = blockIdx.x * 16;
  const int head = blockIdx.y;
  const int tid  = threadIdx.x;
  const int wv   = tid >> 5;
  const int lane = tid & 31;
  const int l15  = lane & 15;
  const float ab = ab_p[0];

  // Async contiguous preload of sj[head][*] into LDS (3 sweeps x 256 lanes x 16B).
  {
    const unsigned base = lds_off(sjh);
    const float* gsrc = sjT + (size_t)head * GN;
#pragma unroll
    for (int j0 = tid * 4; j0 < GN; j0 += 1024) {
      async_copy_b128(base + (unsigned)(j0 * 4), gsrc + j0);
    }
    wait_asynccnt0();
  }
  if (tid < 16) si16[tid] = siT[head * GN + i0 + tid];
  __syncthreads();

  // ---- Phase A: 8 waves x 2 rows, 4 columns per lane per step ----
#pragma unroll
  for (int rr = 0; rr < 2; ++rr) {
    const int r = wv * 2 + rr;
    const int i = i0 + r;
    const int* __restrict__ arow = adj + (size_t)i * GN;
    const float sir = si16[r];

    float m = -3.0e38f;
    for (int jb = lane * 4; jb < GN; jb += 128) {
      const int4   a4 = *(const int4*)(arow + jb);
      const float4 s4 = *(const float4*)(sjh + jb);
      __builtin_prefetch(arow + jb + 512, 0, 0);
      float s0 = sir + s4.x + ab; s0 = (s0 > 0.f) ? s0 : 0.2f * s0;
      float s1 = sir + s4.y + ab; s1 = (s1 > 0.f) ? s1 : 0.2f * s1;
      float s2 = sir + s4.z + ab; s2 = (s2 > 0.f) ? s2 : 0.2f * s2;
      float s3 = sir + s4.w + ab; s3 = (s3 > 0.f) ? s3 : 0.2f * s3;
      s0 = (a4.x == 0) ? -1.0e30f : s0;
      s1 = (a4.y == 0) ? -1.0e30f : s1;
      s2 = (a4.z == 0) ? -1.0e30f : s2;
      s3 = (a4.w == 0) ? -1.0e30f : s3;
      m = fmaxf(m, fmaxf(fmaxf(s0, s1), fmaxf(s2, s3)));
    }
#pragma unroll
    for (int off = 16; off > 0; off >>= 1) m = fmaxf(m, __shfl_xor(m, off, 32));

    float sum = 0.f;
    for (int jb = lane * 4; jb < GN; jb += 128) {
      const int4   a4 = *(const int4*)(arow + jb);
      const float4 s4 = *(const float4*)(sjh + jb);
      float s0 = sir + s4.x + ab; s0 = (s0 > 0.f) ? s0 : 0.2f * s0;
      float s1 = sir + s4.y + ab; s1 = (s1 > 0.f) ? s1 : 0.2f * s1;
      float s2 = sir + s4.z + ab; s2 = (s2 > 0.f) ? s2 : 0.2f * s2;
      float s3 = sir + s4.w + ab; s3 = (s3 > 0.f) ? s3 : 0.2f * s3;
      s0 = (a4.x == 0) ? -1.0e30f : s0;
      s1 = (a4.y == 0) ? -1.0e30f : s1;
      s2 = (a4.z == 0) ? -1.0e30f : s2;
      s3 = (a4.w == 0) ? -1.0e30f : s3;
      const float e0 = __expf(s0 - m);   // all-masked row -> e==1 -> uniform softmax
      const float e1 = __expf(s1 - m);
      const float e2 = __expf(s2 - m);
      const float e3 = __expf(s3 - m);
      sum += (e0 + e1) + (e2 + e3);
      v4h pv;
      pv[0] = (_Float16)e0; pv[1] = (_Float16)e1;
      pv[2] = (_Float16)e2; pv[3] = (_Float16)e3;
      *(v4h*)(probs + r * PROB_STRIDE + jb) = pv;   // ds_store_b64
    }
#pragma unroll
    for (int off = 16; off > 0; off >>= 1) sum += __shfl_xor(sum, off, 32);
    if (lane == 0) rowscale[r] = 1.0f / sum;
  }
  __syncthreads();

  // ---- Phase B: P @ V ----
  const int dt  = wv & 1;        // D tile (0..1)
  const int jc  = wv >> 1;       // K chunk (0..3), each 768 wide
  const int kof = (lane < 16) ? 0 : 8;
  v8f acc = {};
  const _Float16* __restrict__ vcol =
      hT16 + ((size_t)head * GD + dt * 16 + l15) * GN;
  const _Float16* __restrict__ prow = probs + l15 * PROB_STRIDE;

  const int jend = jc * 768 + 768;
  for (int jb = jc * 768; jb < jend; jb += 32) {
    v8h a_lo = *(const v8h*)(prow + jb + kof);
    v8h a_hi = *(const v8h*)(prow + jb + kof + 16);
    v8h b_lo = *(const v8h*)(vcol + jb + kof);
    v8h b_hi = *(const v8h*)(vcol + jb + kof + 16);
    acc = __builtin_amdgcn_wmma_f32_16x16x32_f16(
        false, cat16(a_lo, a_hi), false, cat16(b_lo, b_hi),
        (short)0, acc, false, false);
  }

  float* pp = partial + (dt * 4 + jc) * 256;
  const int mbase = (lane < 16) ? 0 : 8;
#pragma unroll
  for (int r = 0; r < 8; ++r) pp[(mbase + r) * 16 + l15] = acc[r];
  __syncthreads();

  // ---- Epilogue: 16x32 outputs, 2 per thread ----
  for (int idx = tid; idx < 512; idx += 256) {
    const int r = idx >> 5;
    const int d = idx & 31;
    const int dtt = d >> 4, dc = d & 15;
    float v = 0.f;
#pragma unroll
    for (int q = 0; q < 4; ++q) v += partial[((dtt * 4 + q) * 16 + r) * 16 + dc];
    v *= rowscale[r];
    const int i = i0 + r;
    const int o = head * GD + d;
    float res = v + x[(size_t)i * GC + o];                 // identity skip
    res = (res > 0.f) ? res : (__expf(res) - 1.0f);        // ELU
    out[(size_t)i * GC + o] = res;
  }
}

// ---------------------------------------------------------------------------
extern "C" void kernel_launch(void* const* d_in, const int* in_sizes, int n_in,
                              void* d_out, int out_size, void* d_ws, size_t ws_size,
                              hipStream_t stream) {
  const float* x   = (const float*)d_in[0];
  const int*   adj = (const int*)d_in[1];
  const float* W   = (const float*)d_in[2];
  const float* bW  = (const float*)d_in[3];
  const float* a1  = (const float*)d_in[4];
  const float* a2  = (const float*)d_in[5];
  const float* ab  = (const float*)d_in[6];
  float* out = (float*)d_out;

  char* ws = (char*)d_ws;
  const size_t H32_BYTES = (size_t)GN * GC * 4;        // 3.0 MB
  const size_t HT_BYTES  = (size_t)GH * GD * GN * 2;   // 1.5 MB
  float*     h32  = (float*)ws;
  _Float16*  hT16 = (_Float16*)(ws + H32_BYTES);
  float*     siT  = (float*)(ws + H32_BYTES + HT_BYTES);
  float*     sjT  = siT + GN * GH;

  k_gemm_h<<<dim3(GN / 64, GC / 64), 128, 0, stream>>>(x, W, bW, h32, hT16);
  k_scores<<<(GN * GH) / 256, 256, 0, stream>>>(h32, a1, a2, siT, sjT);

  (void)hipFuncSetAttribute((const void*)k_attn,
                            hipFuncAttributeMaxDynamicSharedMemorySize,
                            SMEM_TOTAL);
  k_attn<<<dim3(GN / 16, GH), 256, SMEM_TOTAL, stream>>>(x, adj, siT, sjT, hT16,
                                                         ab, out);
}
